// Confidence_Score_2241972928775
// MI455X (gfx1250) — compile-verified, matrix operand order checked by reading, not run on hardware
//
#include <hip/hip_runtime.h>
#include <hip/hip_bf16.h>
#include <math.h>

// ---------------------------------------------------------------------------
// Problem constants (from the reference)
#define NG     512     // NUM_GRAPHS
#define D_IN   128
#define H_DIM  256

typedef __attribute__((ext_vector_type(16))) __bf16 v16bf;
typedef __attribute__((ext_vector_type(8)))  __bf16 v8bf;
typedef __attribute__((ext_vector_type(8)))  float  v8f;

// ---------------------------------------------------------------------------
// Workspace layout.
// Float region (element offsets in floats):
#define WSF_SUMS   0                          // [NG * D_IN]   f32 segment sums
#define WSF_CNT    (NG * D_IN)                // [NG]          f32 histogram
#define WSF_H2G    (WSF_CNT + NG)             // [NG * H_DIM]  f32 layer-2 out
#define WSF_OUTG   (WSF_H2G + NG * H_DIM)     // [NG]          f32 per-graph out
#define WSF_TOTAL  (WSF_OUTG + NG)            // 197632 floats
// bf16 region starts at byte offset WSF_TOTAL*4 (16B aligned: 790528 % 16 == 0)
#define WSB_SUMS   0                          // [NG * D_IN]   bf16 sums (GEMM1 A)
#define WSB_A2     (NG * D_IN)                // [NG * H_DIM]  bf16 cnt*hG (GEMM2 A)
#define WSB_W1T    (WSB_A2 + NG * H_DIM)      // [H_DIM * D_IN]  W1^T bf16
#define WSB_W2T    (WSB_W1T + H_DIM * D_IN)   // [H_DIM * H_DIM] W2^T bf16

// ---------------------------------------------------------------------------
// Kernel 0: zero the accumulation part of the workspace (sums + counts).
__global__ void zero_ws_kernel(float* __restrict__ ws, int n) {
    int i = blockIdx.x * blockDim.x + threadIdx.x;
    if (i < n) ws[i] = 0.0f;
}

// ---------------------------------------------------------------------------
// Kernel 1: streaming segment-sum over sorted batch ids (the roofline stage:
// ~52 MB of x). One wave owns 128 consecutive rows; lane L reads 4 columns
// (float4 = global_load_b128, one full 512B row per wave step). Run-length
// accumulate in registers while the segment id is constant, flush with
// float atomics (few hundred flushes total across the whole grid).
__global__ void seg_sum_kernel(const float* __restrict__ x,
                               const long long* __restrict__ batch,
                               float* __restrict__ sums,
                               float* __restrict__ counts,
                               int N) {
    const int lane   = threadIdx.x & 31;
    const int waveId = blockIdx.x * (blockDim.x >> 5) + (threadIdx.x >> 5);
    const int ROWS   = 128;
    int r0 = waveId * ROWS;
    if (r0 >= N) return;
    int r1 = r0 + ROWS; if (r1 > N) r1 = N;

    const int col = lane * 4;
    float4 acc = make_float4(0.f, 0.f, 0.f, 0.f);
    float  cnt = 0.f;
    int curSeg = (int)batch[r0];

    for (int r = r0; r < r1; ++r) {
        if (r + 8 < r1)  // keep the stream warm (global_prefetch_b8)
            __builtin_prefetch(x + (size_t)(r + 8) * D_IN + col, 0, 1);
        const int seg = (int)batch[r];
        if (seg != curSeg) {
            float* dst = sums + (size_t)curSeg * D_IN + col;
            atomicAdd(dst + 0, acc.x);
            atomicAdd(dst + 1, acc.y);
            atomicAdd(dst + 2, acc.z);
            atomicAdd(dst + 3, acc.w);
            if (lane == 0) atomicAdd(&counts[curSeg], cnt);
            acc = make_float4(0.f, 0.f, 0.f, 0.f);
            cnt = 0.f;
            curSeg = seg;
        }
        const float4 v = *(const float4*)(x + (size_t)r * D_IN + col);
        acc.x += v.x; acc.y += v.y; acc.z += v.z; acc.w += v.w;
        cnt += 1.f;
    }
    float* dst = sums + (size_t)curSeg * D_IN + col;
    atomicAdd(dst + 0, acc.x);
    atomicAdd(dst + 1, acc.y);
    atomicAdd(dst + 2, acc.z);
    atomicAdd(dst + 3, acc.w);
    if (lane == 0) atomicAdd(&counts[curSeg], cnt);
}

// ---------------------------------------------------------------------------
// Staging: f32 -> bf16 row-major copy (for GEMM1 A).
__global__ void cvt_bf16_kernel(const float* __restrict__ src,
                                __bf16* __restrict__ dst, int n) {
    int i = blockIdx.x * blockDim.x + threadIdx.x;
    if (i < n) dst[i] = (__bf16)src[i];
}

// Staging: f32 [K,Nc] -> bf16 transposed [Nc,K] (weights as K-contiguous rows).
__global__ void cvt_bf16_T_kernel(const float* __restrict__ src, // [K, Nc]
                                  __bf16* __restrict__ dst,      // [Nc, K]
                                  int K, int Nc) {
    int i = blockIdx.x * blockDim.x + threadIdx.x;
    if (i >= K * Nc) return;
    int n = i / K, k = i - n * K;
    dst[i] = (__bf16)src[k * Nc + n];
}

// ---------------------------------------------------------------------------
// Fragment loader. ISA 7.12.2 16-bit A 16x32 (and mirrored B) layout:
//   element e -> k = 16*(e/8) + 8*(lane/16) + (e%8), row = lane%16.
// With K-contiguous bf16 rows, e=0..7 and e=8..15 are each 8 contiguous
// bf16 = 16 bytes => two global_load_b128 per fragment.
__device__ __forceinline__ v16bf load_frag(const __bf16* __restrict__ rowKb,
                                           int lane) {
    const int off = (lane >> 4) << 3;                 // 0 or 8
    const v8bf lo = *(const v8bf*)(rowKb + off);      // k = off .. off+7
    const v8bf hi = *(const v8bf*)(rowKb + off + 16); // k = off+16 .. off+23
    return __builtin_shufflevector(lo, hi, 0, 1, 2, 3, 4, 5, 6, 7,
                                           8, 9, 10, 11, 12, 13, 14, 15);
}

// ---------------------------------------------------------------------------
// Kernel 2: GEMM1 + fusion.  A2 = bf16( cnt[m] * relu(sums @ W1 + b1) ).
// M=512, K=128, N=256; one wave per 16x16 tile; 4 back-to-back WMMAs.
__global__ void gemm1_wmma_kernel(const __bf16* __restrict__ Abf,  // [NG, D_IN]
                                  const __bf16* __restrict__ W1t,  // [H_DIM, D_IN]
                                  const float* __restrict__ b1,    // [H_DIM]
                                  const float* __restrict__ cnt,   // [NG]
                                  __bf16* __restrict__ A2) {       // [NG, H_DIM]
    const int wave = (blockIdx.x * blockDim.x + threadIdx.x) >> 5;
    const int lane = threadIdx.x & 31;
    const int nt = wave & ((H_DIM / 16) - 1);   // 16 N-tiles
    const int mt = wave >> 4;                   // 32 M-tiles
    const int M0 = mt * 16, N0 = nt * 16;
    const __bf16* arow = Abf + (size_t)(M0 + (lane & 15)) * D_IN;
    const __bf16* brow = W1t + (size_t)(N0 + (lane & 15)) * D_IN;
    const int ncol = N0 + (lane & 15);

    v8f acc = {};
#pragma unroll
    for (int kb = 0; kb < D_IN; kb += 32) {
        const v16bf a = load_frag(arow + kb, lane);
        const v16bf b = load_frag(brow + kb, lane);
        acc = __builtin_amdgcn_wmma_f32_16x16x32_bf16(
            false, a, false, b, (short)0, acc, false, false);
    }
    const float bias = b1[ncol];
#pragma unroll
    for (int v = 0; v < 8; ++v) {
        const int m = M0 + v + ((lane >> 4) << 3);
        float val = acc[v] + bias;
        val = val > 0.f ? val : 0.f;
        A2[(size_t)m * H_DIM + ncol] = (__bf16)(cnt[m] * val);
    }
}

// ---------------------------------------------------------------------------
// Kernel 3: GEMM2.  h2G = relu(A2 @ W2 + b2).  M=512, K=256, N=256.
__global__ void gemm2_wmma_kernel(const __bf16* __restrict__ A2,   // [NG, H_DIM]
                                  const __bf16* __restrict__ W2t,  // [H_DIM, H_DIM]
                                  const float* __restrict__ b2,    // [H_DIM]
                                  float* __restrict__ h2G) {       // [NG, H_DIM]
    const int wave = (blockIdx.x * blockDim.x + threadIdx.x) >> 5;
    const int lane = threadIdx.x & 31;
    const int nt = wave & ((H_DIM / 16) - 1);
    const int mt = wave >> 4;
    const int M0 = mt * 16, N0 = nt * 16;
    const __bf16* arow = A2  + (size_t)(M0 + (lane & 15)) * H_DIM;
    const __bf16* brow = W2t + (size_t)(N0 + (lane & 15)) * H_DIM;
    const int ncol = N0 + (lane & 15);

    v8f acc = {};
#pragma unroll
    for (int kb = 0; kb < H_DIM; kb += 32) {
        const v16bf a = load_frag(arow + kb, lane);
        const v16bf b = load_frag(brow + kb, lane);
        acc = __builtin_amdgcn_wmma_f32_16x16x32_bf16(
            false, a, false, b, (short)0, acc, false, false);
    }
    const float bias = b2[ncol];
#pragma unroll
    for (int v = 0; v < 8; ++v) {
        const int m = M0 + v + ((lane >> 4) << 3);
        float val = acc[v] + bias;
        h2G[(size_t)m * H_DIM + ncol] = val > 0.f ? val : 0.f;
    }
}

// ---------------------------------------------------------------------------
// Kernel 4: per-graph confidence head: c = h2G @ Wc + bc; sp/(1+sp).
__global__ void head_kernel(const float* __restrict__ h2G,
                            const float* __restrict__ Wc,
                            const float* __restrict__ bc,
                            float* __restrict__ outG) {
    const int g = blockIdx.x * blockDim.x + threadIdx.x;
    if (g >= NG) return;
    float c = bc[0];
    for (int k = 0; k < H_DIM; ++k) c += h2G[(size_t)g * H_DIM + k] * Wc[k];
    const float sp = (c > 20.f) ? c : log1pf(expf(c));  // softplus
    outG[g] = sp / (1.f + sp);
}

// ---------------------------------------------------------------------------
// Kernel 5: scatter per-graph scalar back to all N nodes.
__global__ void scatter_kernel(const long long* __restrict__ batch,
                               const float* __restrict__ outG,
                               float* __restrict__ out, int N) {
    const int i = blockIdx.x * blockDim.x + threadIdx.x;
    if (i < N) out[i] = outG[(int)batch[i]];
}

// ---------------------------------------------------------------------------
extern "C" void kernel_launch(void* const* d_in, const int* in_sizes, int n_in,
                              void* d_out, int out_size, void* d_ws, size_t ws_size,
                              hipStream_t stream) {
    const float*     x     = (const float*)d_in[0];
    const long long* batch = (const long long*)d_in[1];
    const float*     W1    = (const float*)d_in[2];
    const float*     b1    = (const float*)d_in[3];
    const float*     W2    = (const float*)d_in[4];
    const float*     b2    = (const float*)d_in[5];
    const float*     Wc    = (const float*)d_in[6];
    const float*     bc    = (const float*)d_in[7];
    float*           out   = (float*)d_out;

    const int N = in_sizes[1];           // number of nodes

    float*  wsf  = (float*)d_ws;
    float*  sums = wsf + WSF_SUMS;
    float*  cnt  = wsf + WSF_CNT;
    float*  h2G  = wsf + WSF_H2G;
    float*  outG = wsf + WSF_OUTG;
    __bf16* wsb  = (__bf16*)(wsf + WSF_TOTAL);
    __bf16* sumsB = wsb + WSB_SUMS;
    __bf16* A2    = wsb + WSB_A2;
    __bf16* W1t   = wsb + WSB_W1T;
    __bf16* W2t   = wsb + WSB_W2T;

    // 0) zero accumulators (sums + counts) — every call (ws is not re-poisoned)
    {
        const int n = NG * D_IN + NG;
        zero_ws_kernel<<<(n + 255) / 256, 256, 0, stream>>>(wsf, n);
    }
    // 1) segment sum + histogram (memory-bound stage)
    {
        const int ROWS = 128;
        const int waves = (N + ROWS - 1) / ROWS;
        const int blocks = (waves + 7) / 8;          // 8 waves (256 thr) / block
        seg_sum_kernel<<<blocks, 256, 0, stream>>>(x, batch, sums, cnt, N);
    }
    // 1b) stage bf16 operands: sums (row-major), W1^T and W2^T (K-contiguous)
    cvt_bf16_kernel<<<(NG * D_IN + 255) / 256, 256, 0, stream>>>(sums, sumsB, NG * D_IN);
    cvt_bf16_T_kernel<<<(D_IN * H_DIM + 255) / 256, 256, 0, stream>>>(W1, W1t, D_IN, H_DIM);
    cvt_bf16_T_kernel<<<(H_DIM * H_DIM + 255) / 256, 256, 0, stream>>>(W2, W2t, H_DIM, H_DIM);
    // 2) GEMM1 (fused bias+relu+cnt-scale, bf16 out): 512x128 @ 128x256
    {
        const int waves = (NG / 16) * (H_DIM / 16);  // 512 tiles
        gemm1_wmma_kernel<<<waves / 8, 256, 0, stream>>>(sumsB, W1t, b1, cnt, A2);
    }
    // 3) GEMM2 (fused bias+relu): 512x256 @ 256x256
    {
        const int waves = (NG / 16) * (H_DIM / 16);  // 512 tiles
        gemm2_wmma_kernel<<<waves / 8, 256, 0, stream>>>(A2, W2t, b2, h2G);
    }
    // 4) confidence head per graph
    head_kernel<<<NG / 256, 256, 0, stream>>>(h2G, Wc, bc, outG);
    // 5) scatter to nodes
    scatter_kernel<<<(N + 255) / 256, 256, 0, stream>>>(batch, outG, out, N);
}
